// Siamese_32822140076550
// MI455X (gfx1250) — compile-verified, matrix operand order checked by reading, not run on hardware
//
#include <hip/hip_runtime.h>
#include <stdint.h>

// Siamese GNN matcher on MI455X (gfx1250):
//   g1b/g2b = bf16(g1/g2)            (streaming convert, once)
//   emb_i   = g_ib @ W               (bf16 WMMA, TDM-staged LDS tiles)
//   A0      = relu(emb1 @ emb2^T)    (bf16 WMMA, TDM-staged LDS tiles) -> d_out
//   Sinkhorn via scaling vectors: iterate u <- u/(u*(A0 v)+eps),
//                                 v <- v/(v*(A0^T u)+eps)   (A0 L2-resident)
//   final:  out = diag(u) * A0 * diag(v)  in place.

#define EPSV 0.001f
#define LDS_PITCH 264   // 256 bf16 + 8 bf16 TDM pad (16B per 512B) -> bank spread

typedef __attribute__((ext_vector_type(16))) __bf16 bf16x16;
typedef __attribute__((ext_vector_type(8)))  float  floatx8;
typedef __attribute__((ext_vector_type(4)))  unsigned int uint4v;
typedef __attribute__((ext_vector_type(8)))  int    int8v;
typedef __attribute__((ext_vector_type(4)))  int    int4v;

union FragU {
  bf16x16 v;
  uint4   q[2];
  __bf16  h[16];
};

__device__ __forceinline__ unsigned short bf16_bits(float x) {
  unsigned u = __builtin_bit_cast(unsigned, x);
  u += 0x7FFFu + ((u >> 16) & 1u);                 // round-to-nearest-even
  return (unsigned short)(u >> 16);
}
__device__ __forceinline__ __bf16 bf16_rne(float x) {
  unsigned short hs = bf16_bits(x);
  return __builtin_bit_cast(__bf16, hs);
}

// 16-bit A/B fragment per ISA layout: lane L, row = L%16;
// elements 0..7  <- K = kbase + j       (kbase = k0 + 8*(L>>4))
// elements 8..15 <- K = kbase + 16 + j
__device__ __forceinline__ bf16x16 load_frag(const __bf16* row, int kbase) {
  FragU f;
  f.q[0] = *(const uint4*)(row + kbase);
  f.q[1] = *(const uint4*)(row + kbase + 16);
  return f.v;
}

#define WMMA_BF16(A, Bv, C) \
  __builtin_amdgcn_wmma_f32_16x16x32_bf16(false, (A), false, (Bv), (short)0, (C), false, false)

// ---- TDM: DMA one 64x256 bf16 tile (row stride 256) global -> LDS -----------
// LDS destination gets 16B pad every 512B (pad_interval code 6 = 128 DWORDs,
// pad_amount code 3 = 4 DWORDs) -> effective LDS pitch = 264 bf16 elements.
__device__ __forceinline__ void tdm_load_64x256_bf16(const __bf16* src, unsigned lds_off) {
  uint64_t ga = (uint64_t)(uintptr_t)src;
  uint4v g0;
  g0[0] = 1u;                                   // count=1 (valid user descriptor)
  g0[1] = lds_off;                              // lds_addr (bytes)
  g0[2] = (unsigned)(ga & 0xFFFFFFFFu);         // global_addr[31:0]
  g0[3] = (unsigned)((ga >> 32) & 0x01FFFFFFu)  // global_addr[56:32]
        | (2u << 30);                           // type=2 ("image")
  int8v g1;
  g1[0] = (int)(0x00010000u                     // data_size=1 (2 bytes)
        | (1u << 20)                            // pad_enable
        | (6u << 22)                            // pad_interval: 128 DWORDs
        | (3u << 25));                          // pad_amount:   4 DWORDs
  g1[1] = (int)(256u << 16);                    // tensor_dim0 = 256 (lo16)
  g1[2] = (int)(1024u << 16);                   // tensor_dim0 hi=0 | tensor_dim1 lo = 1024
  g1[3] = (int)(256u << 16);                    // tensor_dim1 hi=0 | tile_dim0 = 256
  g1[4] = 64;                                   // tile_dim1 = 64 | tile_dim2 = 0
  g1[5] = 256;                                  // tensor_dim0_stride = 256 (lo32)
  g1[6] = 0;                                    // stride hi | tensor_dim1_stride lo
  g1[7] = 0;
  int4v z4 = {0, 0, 0, 0};
#if defined(__clang_major__) && __clang_major__ >= 23
  int8v z8 = {0, 0, 0, 0, 0, 0, 0, 0};
  __builtin_amdgcn_tensor_load_to_lds(g0, g1, z4, z4, z8, 0);
#else
  __builtin_amdgcn_tensor_load_to_lds(g0, g1, z4, z4, 0);
#endif
}

// ---- streaming f32 -> bf16 convert (float4 in, 4x bf16 packed out) ----------
__global__ void cvt_bf16_kernel(const float* __restrict__ src, __bf16* __restrict__ dst) {
  int i = blockIdx.x * 256 + threadIdx.x;       // one float4 / thread
  float4 f = ((const float4*)src)[i];
  uint2 o;
  o.x = (unsigned)bf16_bits(f.x) | ((unsigned)bf16_bits(f.y) << 16);
  o.y = (unsigned)bf16_bits(f.z) | ((unsigned)bf16_bits(f.w) << 16);
  ((uint2*)dst)[i] = o;
}

// ---- Wt[e*256+d] = bf16(W[d*256+e]) -----------------------------------------
__global__ void wt_kernel(const float* __restrict__ W, __bf16* __restrict__ Wt) {
  int idx = blockIdx.x * 256 + threadIdx.x;     // 65536 elements
  int e = idx >> 8, d = idx & 255;
  Wt[idx] = bf16_rne(W[d * 256 + e]);
}

// ---- u = v = 1 --------------------------------------------------------------
__global__ void init_uv(float* __restrict__ u, float* __restrict__ v) {
  int i = blockIdx.x * 256 + threadIdx.x;
  u[i] = 1.0f;
  v[i] = 1.0f;
}

// ---- shared GEMM core: C(64x64) += Asm(64x256) * Bsm(64x256)^T --------------
// block = 128 threads (4 waves), wave w -> 32x32 sub-tile (w>>1, w&1).
__device__ __forceinline__ void gemm_tile_from_lds(const __bf16* smA, const __bf16* smB,
                                                   floatx8 acc[2][2]) {
  const int lane = threadIdx.x & 31;
  const int wave = threadIdx.x >> 5;
  const int r    = lane & 15;
  const int hi   = lane >> 4;
  const __bf16* a0row = smA + (size_t)((wave >> 1) * 32 + r) * LDS_PITCH;
  const __bf16* a1row = a0row + (size_t)16 * LDS_PITCH;
  const __bf16* b0row = smB + (size_t)((wave & 1) * 32 + r) * LDS_PITCH;
  const __bf16* b1row = b0row + (size_t)16 * LDS_PITCH;
#pragma unroll
  for (int k0 = 0; k0 < 256; k0 += 32) {
    const int kb = k0 + hi * 8;
    bf16x16 a0 = load_frag(a0row, kb);
    bf16x16 a1 = load_frag(a1row, kb);
    bf16x16 b0 = load_frag(b0row, kb);
    bf16x16 b1 = load_frag(b1row, kb);
    acc[0][0] = WMMA_BF16(a0, b0, acc[0][0]);
    acc[0][1] = WMMA_BF16(a0, b1, acc[0][1]);
    acc[1][0] = WMMA_BF16(a1, b0, acc[1][0]);
    acc[1][1] = WMMA_BF16(a1, b1, acc[1][1]);
  }
}

// ---- emb[16384,256](bf16) = gb[16384,256](bf16) @ W (via Wt rows) -----------
// grid (N/64=4, Mtot/64=256), block 128.
__global__ void proj_kernel(const __bf16* __restrict__ gb, const __bf16* __restrict__ Wt,
                            __bf16* __restrict__ emb) {
  __shared__ __bf16 smA[64 * LDS_PITCH];
  __shared__ __bf16 smB[64 * LDS_PITCH];
  const int wave = threadIdx.x >> 5;
  const int lane = threadIdx.x & 31;
  const int r = lane & 15, hi = lane >> 4;
  const int m0 = blockIdx.y * 64;
  const int n0 = blockIdx.x * 64;

  if (wave == 0) {
    tdm_load_64x256_bf16(gb + (size_t)m0 * 256, (unsigned)(uintptr_t)smA);
    tdm_load_64x256_bf16(Wt + (size_t)n0 * 256, (unsigned)(uintptr_t)smB);
    __builtin_amdgcn_s_wait_tensorcnt(0);
  }
  __syncthreads();

  floatx8 acc[2][2] = {};
  gemm_tile_from_lds(smA, smB, acc);

  // C layout: element e of v8f at lane L -> row +8*hi+e, col +(L&15)
#pragma unroll
  for (int mi = 0; mi < 2; ++mi)
#pragma unroll
    for (int ni = 0; ni < 2; ++ni) {
      const int col = n0 + (wave & 1) * 32 + ni * 16 + r;
#pragma unroll
      for (int e = 0; e < 8; ++e) {
        const int rowm = m0 + (wave >> 1) * 32 + mi * 16 + hi * 8 + e;
        emb[(size_t)rowm * 256 + col] = bf16_rne(acc[mi][ni][e]);
      }
    }
}

// ---- A0[b] = relu(emb1[b] @ emb2[b]^T) -> f32 in d_out ----------------------
// grid (16,16,16) = (ntile64, mtile64, batch), block 128.
__global__ void bmm_kernel(const __bf16* __restrict__ emb1, const __bf16* __restrict__ emb2,
                           float* __restrict__ A) {
  __shared__ __bf16 smA[64 * LDS_PITCH];
  __shared__ __bf16 smB[64 * LDS_PITCH];
  const int wave = threadIdx.x >> 5;
  const int lane = threadIdx.x & 31;
  const int r = lane & 15, hi = lane >> 4;
  const int b  = blockIdx.z;
  const int m0 = blockIdx.y * 64;
  const int n0 = blockIdx.x * 64;
  const __bf16* e1 = emb1 + ((size_t)b << 18);   // 1024*256
  const __bf16* e2 = emb2 + ((size_t)b << 18);

  if (wave == 0) {
    tdm_load_64x256_bf16(e1 + (size_t)m0 * 256, (unsigned)(uintptr_t)smA);
    tdm_load_64x256_bf16(e2 + (size_t)n0 * 256, (unsigned)(uintptr_t)smB);
    __builtin_amdgcn_s_wait_tensorcnt(0);
  }
  __syncthreads();

  floatx8 acc[2][2] = {};
  gemm_tile_from_lds(smA, smB, acc);

  float* Ab = A + ((size_t)b << 20);             // 1024*1024
#pragma unroll
  for (int mi = 0; mi < 2; ++mi)
#pragma unroll
    for (int ni = 0; ni < 2; ++ni) {
      const int col = n0 + (wave & 1) * 32 + ni * 16 + r;
#pragma unroll
      for (int e = 0; e < 8; ++e) {
        const int rowm = m0 + (wave >> 1) * 32 + mi * 16 + hi * 8 + e;
        float vv = acc[mi][ni][e];
        Ab[(size_t)rowm * 1024 + col] = vv > 0.0f ? vv : 0.0f;
      }
    }
}

// ---- row pass: y_m = (A0 v)_m ; u_m <- u_m / (u_m*y_m + EPS) ----------------
// grid (1024, 16) = (m, b), block 256; float4 streaming loads.
__global__ void rowpass(const float* __restrict__ A0, const float* __restrict__ v,
                        float* __restrict__ u) {
  const int m = blockIdx.x, b = blockIdx.y, t = threadIdx.x;
  const float4* row = (const float4*)(A0 + ((size_t)b << 20) + ((size_t)m << 10));
  const float4* vb  = (const float4*)(v + (b << 10));
  float4 a = row[t];
  float4 w = vb[t];
  float s = a.x * w.x + a.y * w.y + a.z * w.z + a.w * w.w;
  __shared__ float red[256];
  red[t] = s;
  __syncthreads();
  for (int off = 128; off > 0; off >>= 1) {
    if (t < off) red[t] += red[t + off];
    __syncthreads();
  }
  if (t == 0) {
    const int i = (b << 10) + m;
    float uu = u[i];
    u[i] = uu / (uu * red[0] + EPSV);
  }
}

// ---- col pass: z_n = (A0^T u)_n ; v_n <- v_n / (v_n*z_n + EPS) --------------
// grid (4, 16) = (ntile256, b), block 1024 (256 cols x 4 m-segments).
__global__ void colpass(const float* __restrict__ A0, const float* __restrict__ u,
                        float* __restrict__ v) {
  const int b   = blockIdx.y;
  const int nl  = threadIdx.x & 255;
  const int seg = threadIdx.x >> 8;
  const int n   = (blockIdx.x << 8) + nl;
  const float* ub   = u + (b << 10);
  const float* base = A0 + ((size_t)b << 20) + n;
  float s = 0.0f;
  const int mstart = seg << 8;
  for (int mm = mstart; mm < mstart + 256; ++mm)
    s += base[(size_t)mm << 10] * ub[mm];
  __shared__ float red[4][256];
  red[seg][nl] = s;
  __syncthreads();
  if (seg == 0) {
    float z = red[0][nl] + red[1][nl] + red[2][nl] + red[3][nl];
    const int i = (b << 10) + n;
    float vv = v[i];
    v[i] = vv / (vv * z + EPSV);
  }
}

// ---- final: A *= u_m * v_n (in place in d_out) ------------------------------
__global__ void finalize_kernel(float* __restrict__ A, const float* __restrict__ u,
                                const float* __restrict__ v) {
  const size_t idx = (size_t)blockIdx.x * 256 + threadIdx.x;
  const int b = (int)(idx >> 20);
  const int m = (int)(idx >> 10) & 1023;
  const int n = (int)idx & 1023;
  A[idx] = A[idx] * u[(b << 10) + m] * v[(b << 10) + n];
}

extern "C" void kernel_launch(void* const* d_in, const int* in_sizes, int n_in,
                              void* d_out, int out_size, void* d_ws, size_t ws_size,
                              hipStream_t stream) {
  const float* g1 = (const float*)d_in[0];   // [16,1024,256] f32
  const float* g2 = (const float*)d_in[1];   // [16,1024,256] f32
  const float* W  = (const float*)d_in[2];   // [256,256]     f32
  float* out = (float*)d_out;                // [16,1024,1024] f32; holds A0

  // workspace layout (~32.4 MB)
  char* ws = (char*)d_ws;
  __bf16* g1b  = (__bf16*)ws;                               // 8 MB
  __bf16* g2b  = (__bf16*)(ws + (size_t)( 8u << 20));       // 8 MB
  __bf16* emb1 = (__bf16*)(ws + (size_t)(16u << 20));       // 8 MB
  __bf16* emb2 = (__bf16*)(ws + (size_t)(24u << 20));       // 8 MB
  __bf16* Wt   = (__bf16*)(ws + (size_t)(32u << 20));       // 128 KB
  float*  u    = (float*)(ws + (size_t)(32u << 20) + (size_t)(128u << 10));
  float*  v    = u + 16384;

  cvt_bf16_kernel<<<4096, 256, 0, stream>>>(g1, g1b);       // 4.19M float4s /4
  cvt_bf16_kernel<<<4096, 256, 0, stream>>>(g2, g2b);
  wt_kernel<<<256, 256, 0, stream>>>(W, Wt);
  init_uv<<<64, 256, 0, stream>>>(u, v);

  proj_kernel<<<dim3(4, 256), 128, 0, stream>>>(g1b, Wt, emb1);
  proj_kernel<<<dim3(4, 256), 128, 0, stream>>>(g2b, Wt, emb2);

  bmm_kernel<<<dim3(16, 16, 16), 128, 0, stream>>>(emb1, emb2, out);

  for (int it = 0; it < 10; ++it) {
    rowpass<<<dim3(1024, 16), 256, 0, stream>>>(out, v, u);
    colpass<<<dim3(4, 16), 1024, 0, stream>>>(out, u, v);
  }

  finalize_kernel<<<65536, 256, 0, stream>>>(out, u, v);
}
